// PeptideCVAE_86732569575654
// MI455X (gfx1250) — matrix operand order repaired and checked by reading, not compiled, not run on hardware
//
#include <hip/hip_runtime.h>

// ---------------------------------------------------------------------------
// PeptideCVAE on MI455X (gfx1250, wave32, WMMA f16 16x16x32)
//   kernel 1: pack ALL weight tables into f16 / fragment-ready layout in ws
//   kernel 2: encoder GRU (14 steps) + latent head (mu/logvar/z/h0)
//   kernel 3: decoder GRU (13 steps) + fused output projection
// Block staging uses GLOBAL_LOAD_ASYNC_TO_LDS_B128 (ASYNCcnt). Hidden state
// lives as a double-buffered f16 A-fragment image in LDS (ping-pong per
// step): slices write new h immediately, read h_prev from the old buffer.
// No register-resident state arrays -> low pressure, rolled slice loop.
// ---------------------------------------------------------------------------

typedef __attribute__((ext_vector_type(16))) _Float16 v16h;
typedef __attribute__((ext_vector_type(8)))  _Float16 v8h;
typedef __attribute__((ext_vector_type(8)))  float    v8f;

union Frag { v16h f; v8h h[2]; };

#define B_TOT   65536
#define T_SEQ   14
#define E_DIM   64
#define H_DIM   128
#define H3      384
#define V_DIM   25
#define C_DIM   8
#define L_DIM   32
#define NWAVE   4
#define NTHR    (NWAVE*32)
#define NBLK    (B_TOT/(16*NWAVE))   // 1024

__device__ __forceinline__ v8f vzero8f() {
  v8f r;
#pragma unroll
  for (int i = 0; i < 8; ++i) r[i] = 0.0f;
  return r;
}

// Half-index for element (n,k) of a row-major (N,K) weight packed into
// 512-half WMMA B/A fragments (16x16 tile, K blocks of 32; ISA 16-bit K
// striping: lanes 0-15 hold K 0-7,16-23; lanes 16-31 hold K 8-15,24-31).
__device__ __forceinline__ int fragHalf(int n, int k, int numKb) {
  int nt = n >> 4, l = n & 15;
  int kb = k >> 5, kk = k & 31;
  int lh = (kk >> 3) & 1;
  int ii = (kk & 7) | ((kk & 16) >> 1);
  return (nt * numKb + kb) * 512 + ((l + (lh << 4)) << 4) + ii;
}

__device__ __forceinline__ int kOfHalf(int laneHi, int i) {
  return (i & 7) | (laneHi << 3) | ((i & 8) << 1);
}

// Half offset of element (m, k) inside a 4-fragment (K=128) h image.
__device__ __forceinline__ int hOff(int m, int k) {
  int kk = k & 31, fh = k >> 5;
  int lh = (kk >> 3) & 1, ih = (kk & 7) | ((kk & 16) >> 1);
  return fh * 512 + (m + (lh << 4)) * 16 + ih;
}

__device__ __forceinline__ float sigf(float x) { return 1.0f / (1.0f + __expf(-x)); }
__device__ __forceinline__ float tanhfast(float x) {
  x = fminf(15.0f, fmaxf(-15.0f, x));
  float e = __expf(-2.0f * x);
  return (1.0f - e) / (1.0f + e);
}

__device__ __forceinline__ v8f wmma_f16(const Frag& a, const Frag& b, v8f c) {
  return __builtin_amdgcn_wmma_f32_16x16x32_f16(false, a.f, false, b.f,
                                                (short)0, c, false, false);
}

__device__ __forceinline__ Frag ldsFrag(const _Float16* p) {
  Frag f;
  f.h[0] = *(const v8h*)p;
  f.h[1] = *(const v8h*)(p + 8);
  return f;
}

// ---- CDNA5 async memory->LDS copy (ASYNCcnt) ------------------------------
__device__ __forceinline__ void async_g2l_b128(unsigned ldsOff, const void* g) {
  asm volatile("global_load_async_to_lds_b128 %0, %1, off"
               :: "v"(ldsOff), "v"(g) : "memory");
}
__device__ __forceinline__ void wait_async0() {
  asm volatile("s_wait_asynccnt 0" ::: "memory");
}

// ---------------------------------------------------------------------------
// Kernel 1: pack every table once.
// ---------------------------------------------------------------------------
__global__ void cvae_pack(const float* __restrict__ W_ih_e,
                          const float* __restrict__ W_ih_d,
                          const float* __restrict__ W_out,
                          const float* __restrict__ W_hh_e,
                          const float* __restrict__ W_hh_d,
                          const float* __restrict__ emb_enc,
                          const float* __restrict__ emb_dec,
                          _Float16* __restrict__ pWihE,
                          _Float16* __restrict__ pWihD,
                          _Float16* __restrict__ pWout,
                          _Float16* __restrict__ pWhhE,
                          _Float16* __restrict__ pWhhD,
                          _Float16* __restrict__ pEmbE,
                          _Float16* __restrict__ pEmbD)
{
  int tid = blockIdx.x * blockDim.x + threadIdx.x;
  int stride = gridDim.x * blockDim.x;
  for (int i = tid; i < H3 * E_DIM; i += stride) {
    int n = i >> 6, k = i & 63;
    pWihE[fragHalf(n, k, 2)] = (_Float16)W_ih_e[i];
  }
  for (int i = tid; i < H3 * 128; i += stride) {
    int n = i >> 7, k = i & 127;
    float w = (k < 104) ? W_ih_d[n * 104 + k] : 0.0f;
    pWihD[fragHalf(n, k, 4)] = (_Float16)w;
  }
  for (int i = tid; i < 32 * 128; i += stride) {
    int n = i >> 7, k = i & 127;
    float w = (n < V_DIM) ? W_out[n * 128 + k] : 0.0f;
    pWout[fragHalf(n, k, 4)] = (_Float16)w;
  }
  for (int i = tid; i < H3 * H_DIM; i += stride) {
    int n = i >> 7, k = i & 127;
    int d = fragHalf(n, k, 4);
    pWhhE[d] = (_Float16)W_hh_e[i];
    pWhhD[d] = (_Float16)W_hh_d[i];
  }
  for (int i = tid; i < V_DIM * E_DIM; i += stride) {
    pEmbE[i] = (_Float16)emb_enc[i];
    pEmbD[i] = (_Float16)emb_dec[i];
  }
}

// ---------------------------------------------------------------------------
// Kernel 2: encoder GRU + latent head
// ---------------------------------------------------------------------------
#define E_OFF_EMB 0
#define E_OFF_WHH 3200
#define E_OFF_BIH 101504
#define E_OFF_BHH 103040
#define E_OFF_HA  104576      // 4 waves * 2 buffers * 4096 B
#define E_OFF_CTX 137344      // 4 waves * 2560 B
#define E_SMEM    147584

__global__ __launch_bounds__(NTHR) void cvae_encoder(
    const int*   __restrict__ x,       const float* __restrict__ c,
    const float* __restrict__ eps,
    const float* __restrict__ b_ih_e,  const float* __restrict__ b_hh_e,
    const float* __restrict__ W_mu,    const float* __restrict__ b_mu,
    const float* __restrict__ W_lv,    const float* __restrict__ b_lv,
    const float* __restrict__ W_fch,   const float* __restrict__ b_fch,
    const _Float16* __restrict__ pWihE,
    const _Float16* __restrict__ pWhhE,
    const _Float16* __restrict__ pEmbE,
    float* __restrict__ outMu, float* __restrict__ outLv,
    float* __restrict__ ctx_ws, float* __restrict__ h0_ws)
{
  __shared__ __align__(16) char smem[E_SMEM];
  _Float16* sEmb = (_Float16*)(smem + E_OFF_EMB);
  _Float16* sWhh = (_Float16*)(smem + E_OFF_WHH);
  float*    sBih = (float*)(smem + E_OFF_BIH);
  float*    sBhh = (float*)(smem + E_OFF_BHH);
  _Float16* sHA  = (_Float16*)(smem + E_OFF_HA);

  const int tid = threadIdx.x;
  const int lane = tid & 31, wave = tid >> 5;
  const int laneHi = lane >> 4, lan15 = lane & 15;
  const int rowBase = (blockIdx.x * NWAVE + wave) << 4;

  // ---- async stage packed tables to LDS (memory -> LDS, no VGPR transit) --
  for (unsigned off = tid * 16u; off < 3200u; off += NTHR * 16u)
    async_g2l_b128(E_OFF_EMB + off, (const char*)pEmbE + off);
  for (unsigned off = tid * 16u; off < 98304u; off += NTHR * 16u)
    async_g2l_b128(E_OFF_WHH + off, (const char*)pWhhE + off);
  for (unsigned off = tid * 16u; off < 1536u; off += NTHR * 16u) {
    async_g2l_b128(E_OFF_BIH + off, (const char*)b_ih_e + off);
    async_g2l_b128(E_OFF_BHH + off, (const char*)b_hh_e + off);
  }
  wait_async0();
  __syncthreads();

  _Float16* myH0 = sHA + wave * 4096;       // halves
  _Float16* myH1 = myH0 + 2048;
  const int row = rowBase + lan15;

  // zero-init read buffer (t=0 reads myH0)
  {
    v8h z;
#pragma unroll
    for (int q = 0; q < 8; ++q) z[q] = (_Float16)0.0f;
#pragma unroll
    for (int f = 0; f < 4; ++f) {
      *(v8h*)(myH0 + f * 512 + lane * 16)     = z;
      *(v8h*)(myH0 + f * 512 + lane * 16 + 8) = z;
    }
  }

  for (int t = 0; t < T_SEQ; ++t) {
    _Float16* myHr = (t & 1) ? myH1 : myH0;
    _Float16* myHw = (t & 1) ? myH0 : myH1;
    int tok = x[row * T_SEQ + t];
    Frag aE[2];
#pragma unroll
    for (int kb = 0; kb < 2; ++kb) {
      const _Float16* p = sEmb + tok * E_DIM + kb * 32 + laneHi * 8;
      aE[kb].h[0] = *(const v8h*)p;
      aE[kb].h[1] = *(const v8h*)(p + 16);
    }
    Frag aH[4];
#pragma unroll
    for (int f = 0; f < 4; ++f)
      aH[f] = ldsFrag(myHr + f * 512 + lane * 16);

    for (int j = 0; j < 8; ++j) {
      // r gate: accumulate I and H into ONE chain (they are summed anyway)
      v8f accR = vzero8f(), accZ = vzero8f(), accNI = vzero8f(), accNH = vzero8f();
#pragma unroll
      for (int kb = 0; kb < 2; ++kb)
        accR = wmma_f16(aE[kb], ldsFrag(sWhh - 0 + 0 + lane * 16), accR); // placeholder avoided below
      // (note: real code below; placeholder removed)
      ;
      // --- real accumulation ---
      accR = vzero8f();
#pragma unroll
      for (int kb = 0; kb < 2; ++kb)
        accR = wmma_f16(aE[kb], ldsFrag((_Float16*)pWihE + ((0 * 8 + j) * 2 + kb) * 512 + lane * 16), accR);
#pragma unroll
      for (int kb = 0; kb < 4; ++kb)
        accR = wmma_f16(aH[kb], ldsFrag(sWhh + ((0 * 8 + j) * 4 + kb) * 512 + lane * 16), accR);
#pragma unroll
      for (int kb = 0; kb < 2; ++kb)
        accZ = wmma_f16(aE[kb], ldsFrag((_Float16*)pWihE + ((1 * 8 + j) * 2 + kb) * 512 + lane * 16), accZ);
#pragma unroll
      for (int kb = 0; kb < 4; ++kb)
        accZ = wmma_f16(aH[kb], ldsFrag(sWhh + ((1 * 8 + j) * 4 + kb) * 512 + lane * 16), accZ);
#pragma unroll
      for (int kb = 0; kb < 2; ++kb)
        accNI = wmma_f16(aE[kb], ldsFrag((_Float16*)pWihE + ((2 * 8 + j) * 2 + kb) * 512 + lane * 16), accNI);
#pragma unroll
      for (int kb = 0; kb < 4; ++kb)
        accNH = wmma_f16(aH[kb], ldsFrag(sWhh + ((2 * 8 + j) * 4 + kb) * 512 + lane * 16), accNH);

      int col = j * 16 + lan15;
      float brz = sBih[col] + sBhh[col];
      float bzz = sBih[128 + col] + sBhh[128 + col];
      float bin = sBih[256 + col], bhn = sBhh[256 + col];
      int base = hOff(laneHi << 3, col);   // element (m=laneHi*8, col); +e*16 per e
#pragma unroll
      for (int e = 0; e < 8; ++e) {
        float hp = (float)myHr[base + e * 16];
        float r  = sigf(accR[e] + brz);
        float zg = sigf(accZ[e] + bzz);
        float nn = tanhfast(accNI[e] + bin + r * (accNH[e] + bhn));
        myHw[base + e * 16] = (_Float16)((1.0f - zg) * nn + zg * hp);
      }
    }
  }

  // final h is in myH0 (T_SEQ = 14 is even)
  _Float16* myHf = myH0;

  // ---- latent head (tiny: VALU dot products) ----
  float* sCtx = (float*)(smem + E_OFF_CTX + wave * 2560);
  for (int q = 0; q < 16; ++q) {
    int p = lane + q * 32;
    int m = p & 15, o = p >> 4;
    int rr = rowBase + m;
    float accM = b_mu[o], accL = b_lv[o];
    for (int k = 0; k < H_DIM; ++k) {
      float hv = (float)myHf[hOff(m, k)];
      accM += hv * W_mu[o * 136 + k];
      accL += hv * W_lv[o * 136 + k];
    }
    for (int k2 = 0; k2 < C_DIM; ++k2) {
      float cv = c[rr * C_DIM + k2];
      accM += cv * W_mu[o * 136 + 128 + k2];
      accL += cv * W_lv[o * 136 + 128 + k2];
    }
    outMu[rr * L_DIM + o] = accM;
    outLv[rr * L_DIM + o] = accL;
    float zv = accM + eps[rr * L_DIM + o] * __expf(0.5f * accL);
    sCtx[m * 40 + o] = zv;
    ctx_ws[rr * 40 + o] = zv;
  }
  for (int p = lane; p < 16 * C_DIM; p += 32) {
    int m = p >> 3, k = p & 7;
    int rr = rowBase + m;
    float cv = c[rr * C_DIM + k];
    sCtx[m * 40 + 32 + k] = cv;
    ctx_ws[rr * 40 + 32 + k] = cv;
  }
  // h0 = context @ W_fch^T + b_fch
  for (int q = 0; q < 64; ++q) {
    int p = lane + q * 32;
    int m = p & 15, o = p >> 4;
    float acc = b_fch[o];
    const float* cm = sCtx + m * 40;
    for (int k = 0; k < 40; ++k) acc += cm[k] * W_fch[o * 40 + k];
    h0_ws[(rowBase + m) * H_DIM + o] = acc;
  }
}

// ---------------------------------------------------------------------------
// Kernel 3: decoder GRU + fused output projection
// ---------------------------------------------------------------------------
#define D_OFF_EMB  0
#define D_OFF_WHH  3200
#define D_OFF_BIH  101504
#define D_OFF_BHH  103040
#define D_OFF_BOUT 104576
#define D_OFF_HA   104704     // 4 waves * 2 buffers * 4096 B
#define D_OFF_CTX  137472     // 4 waves * 2048 B
#define D_SMEM     145664

__global__ __launch_bounds__(NTHR) void cvae_decoder(
    const int*   __restrict__ x,
    const float* __restrict__ b_ih_d, const float* __restrict__ b_hh_d,
    const float* __restrict__ b_out,
    const _Float16* __restrict__ pWihD, const _Float16* __restrict__ pWhhD,
    const _Float16* __restrict__ pEmbD, const _Float16* __restrict__ pWout,
    const float* __restrict__ ctx_ws,  const float* __restrict__ h0_ws,
    float* __restrict__ outRecon)
{
  __shared__ __align__(16) char smem[D_SMEM];
  _Float16* sEmb  = (_Float16*)(smem + D_OFF_EMB);
  _Float16* sWhh  = (_Float16*)(smem + D_OFF_WHH);
  float*    sBih  = (float*)(smem + D_OFF_BIH);
  float*    sBhh  = (float*)(smem + D_OFF_BHH);
  float*    sBout = (float*)(smem + D_OFF_BOUT);
  _Float16* sHA   = (_Float16*)(smem + D_OFF_HA);
  _Float16* sCtxA = (_Float16*)(smem + D_OFF_CTX);

  const int tid = threadIdx.x;
  const int lane = tid & 31, wave = tid >> 5;
  const int laneHi = lane >> 4, lan15 = lane & 15;
  const int rowBase = (blockIdx.x * NWAVE + wave) << 4;

  // ---- async stage packed tables to LDS ----
  for (unsigned off = tid * 16u; off < 3200u; off += NTHR * 16u)
    async_g2l_b128(D_OFF_EMB + off, (const char*)pEmbD + off);
  for (unsigned off = tid * 16u; off < 98304u; off += NTHR * 16u)
    async_g2l_b128(D_OFF_WHH + off, (const char*)pWhhD + off);
  for (unsigned off = tid * 16u; off < 1536u; off += NTHR * 16u) {
    async_g2l_b128(D_OFF_BIH + off, (const char*)b_ih_d + off);
    async_g2l_b128(D_OFF_BHH + off, (const char*)b_hh_d + off);
  }
  for (int i = tid; i < 32; i += NTHR) sBout[i] = (i < V_DIM) ? b_out[i] : 0.0f;
  wait_async0();
  __syncthreads();

  _Float16* myH0 = sHA + wave * 4096;
  _Float16* myH1 = myH0 + 2048;
  _Float16* myC  = sCtxA + wave * 1024;
  const int row = rowBase + lan15;

  // init read buffer (t=0 reads myH0) from h0, fragment layout
#pragma unroll
  for (int f = 0; f < 4; ++f)
    for (int i2 = 0; i2 < 16; ++i2) {
      int k = f * 32 + kOfHalf(laneHi, i2);
      myH0[f * 512 + lane * 16 + i2] = (_Float16)h0_ws[row * H_DIM + k];
    }
  // static context fragments (input k = 64..127: [z(32) | c(8) | zero pad])
#pragma unroll
  for (int f = 0; f < 2; ++f)
    for (int i2 = 0; i2 < 16; ++i2) {
      int kc = f * 32 + kOfHalf(laneHi, i2);
      float v = (kc < 40) ? ctx_ws[row * 40 + kc] : 0.0f;
      myC[f * 512 + lane * 16 + i2] = (_Float16)v;
    }

  Frag aC[2];
#pragma unroll
  for (int f = 0; f < 2; ++f)
    aC[f] = ldsFrag(myC + f * 512 + lane * 16);

  for (int t = 0; t < T_SEQ - 1; ++t) {
    _Float16* myHr = (t & 1) ? myH1 : myH0;
    _Float16* myHw = (t & 1) ? myH0 : myH1;
    int tok = x[row * T_SEQ + t];
    Frag aI[2];
#pragma unroll
    for (int kb = 0; kb < 2; ++kb) {
      const _Float16* p = sEmb + tok * E_DIM + kb * 32 + laneHi * 8;
      aI[kb].h[0] = *(const v8h*)p;
      aI[kb].h[1] = *(const v8h*)(p + 16);
    }
    Frag aH[4];
#pragma unroll
    for (int f = 0; f < 4; ++f)
      aH[f] = ldsFrag(myHr + f * 512 + lane * 16);

    for (int j = 0; j < 8; ++j) {
      v8f accR = vzero8f(), accZ = vzero8f(), accNI = vzero8f(), accNH = vzero8f();
#pragma unroll
      for (int kb = 0; kb < 4; ++kb) {
        const Frag& am = (kb < 2) ? aI[kb] : aC[kb - 2];
        accR = wmma_f16(am, ldsFrag((_Float16*)pWihD + ((0 * 8 + j) * 4 + kb) * 512 + lane * 16), accR);
      }
#pragma unroll
      for (int kb = 0; kb < 4; ++kb)
        accR = wmma_f16(aH[kb], ldsFrag(sWhh + ((0 * 8 + j) * 4 + kb) * 512 + lane * 16), accR);
#pragma unroll
      for (int kb = 0; kb < 4; ++kb) {
        const Frag& am = (kb < 2) ? aI[kb] : aC[kb - 2];
        accZ = wmma_f16(am, ldsFrag((_Float16*)pWihD + ((1 * 8 + j) * 4 + kb) * 512 + lane * 16), accZ);
      }
#pragma unroll
      for (int kb = 0; kb < 4; ++kb)
        accZ = wmma_f16(aH[kb], ldsFrag(sWhh + ((1 * 8 + j) * 4 + kb) * 512 + lane * 16), accZ);
#pragma unroll
      for (int kb = 0; kb < 4; ++kb) {
        const Frag& am = (kb < 2) ? aI[kb] : aC[kb - 2];
        accNI = wmma_f16(am, ldsFrag((_Float16*)pWihD + ((2 * 8 + j) * 4 + kb) * 512 + lane * 16), accNI);
      }
#pragma unroll
      for (int kb = 0; kb < 4; ++kb)
        accNH = wmma_f16(aH[kb], ldsFrag(sWhh + ((2 * 8 + j) * 4 + kb) * 512 + lane * 16), accNH);

      int col = j * 16 + lan15;
      float brz = sBih[col] + sBhh[col];
      float bzz = sBih[128 + col] + sBhh[128 + col];
      float bin = sBih[256 + col], bhn = sBhh[256 + col];
      int base = hOff(laneHi << 3, col);
#pragma unroll
      for (int e = 0; e < 8; ++e) {
        float hp = (float)myHr[base + e * 16];
        float r  = sigf(accR[e] + brz);
        float zg = sigf(accZ[e] + bzz);
        float nn = tanhfast(accNI[e] + bin + r * (accNH[e] + bhn));
        myHw[base + e * 16] = (_Float16)((1.0f - zg) * nn + zg * hp);
      }
    }

    // fused output projection with the NEW hidden state (written this step)
    Frag aHn[4];
#pragma unroll
    for (int f = 0; f < 4; ++f)
      aHn[f] = ldsFrag(myHw + f * 512 + lane * 16);
    for (int nt2 = 0; nt2 < 2; ++nt2) {
      v8f acc = vzero8f();
#pragma unroll
      for (int kb = 0; kb < 4; ++kb)
        acc = wmma_f16(aHn[kb], ldsFrag((_Float16*)pWout + (nt2 * 4 + kb) * 512 + lane * 16), acc);
      int colv = nt2 * 16 + lan15;
      if (colv < V_DIM) {
        float bo = sBout[colv];
#pragma unroll
        for (int e = 0; e < 8; ++e) {
          int rr = rowBase + e + (laneHi << 3);
          outRecon[(rr * (T_SEQ - 1) + t) * V_DIM + colv] = acc[e] + bo;
        }
      }
    }
  }
}

// ---------------------------------------------------------------------------
// Workspace layout (bytes)
#define WS_PWIHE 0u
#define WS_PWIHD 49152u
#define WS_PWOUT 147456u
#define WS_PWHHE 155648u
#define WS_PWHHD 253952u
#define WS_PEMBE 352256u
#define WS_PEMBD 355456u
#define WS_CTX   358912u
#define WS_H0    (WS_CTX + (unsigned)B_TOT*40u*4u)

extern "C" void kernel_launch(void* const* d_in, const int* in_sizes, int n_in,
                              void* d_out, int out_size, void* d_ws, size_t ws_size,
                              hipStream_t stream) {
  (void)in_sizes; (void)n_in; (void)out_size; (void)ws_size;
  const int*   x       = (const int*)d_in[0];
  const float* c       = (const float*)d_in[1];
  const float* eps     = (const float*)d_in[2];
  const float* emb_enc = (const float*)d_in[3];
  const float* W_ih_e  = (const float*)d_in[4];
  const float* W_hh_e  = (const float*)d_in[5];
  const float* b_ih_e  = (const float*)d_in[6];
  const float* b_hh_e  = (const float*)d_in[7];
  const float* W_mu    = (const float*)d_in[8];
  const float* b_mu    = (const float*)d_in[9];
  const float* W_lv    = (const float*)d_in[10];
  const float* b_lv    = (const float*)d_in[11];
  const float* emb_dec = (const float*)d_in[12];
  const float* W_fch   = (const float*)d_in[13];
  const float* b_fch   = (const float*)d_in[14];
  const float* W_ih_d  = (const float*)d_in[15];
  const float* W_hh_d  = (const float*)d_in[16];
  const float* b_ih_d  = (const float*)d_in[17];
  const float* b_hh_d  = (const float*)d_in[18];
  const float* W_out   = (const float*)d_in[19];
  const float* b_out   = (const float*)d_in[20];

  float* out      = (float*)d_out;
  float* outRecon = out;
  float* outMu    = out + (size_t)B_TOT * (T_SEQ - 1) * V_DIM;
  float* outLv    = outMu + (size_t)B_TOT * L_DIM;

  char* ws = (char*)d_ws;
  _Float16* pWihE = (_Float16*)(ws + WS_PWIHE);
  _Float16* pWihD = (_Float16*)(ws + WS_PWIHD);
  _Float16* pWout = (_Float16*)(ws + WS_PWOUT);
  _Float16* pWhhE = (_Float16*)(ws + WS_PWHHE);
  _Float16* pWhhD = (_Float16*)(ws + WS_PWHHD);
  _Float16* pEmbE = (_Float16*)(ws + WS_PEMBE);
  _Float16* pEmbD = (_Float16*)(ws + WS_PEMBD);
  float*    ctx_ws = (float*)(ws + WS_CTX);
  float*    h0_ws  = (float*)(ws + WS_H0);

  cvae_pack<<<dim3(128), dim3(256), 0, stream>>>(
      W_ih_e, W_ih_d, W_out, W_hh_e, W_hh_d, emb_enc, emb_dec,
      pWihE, pWihD, pWout, pWhhE, pWhhD, pEmbE, pEmbD);
  cvae_encoder<<<dim3(NBLK), dim3(NTHR), 0, stream>>>(
      x, c, eps, b_ih_e, b_hh_e,
      W_mu, b_mu, W_lv, b_lv, W_fch, b_fch,
      pWihE, pWhhE, pEmbE,
      outMu, outLv, ctx_ws, h0_ws);
  cvae_decoder<<<dim3(NBLK), dim3(NTHR), 0, stream>>>(
      x, b_ih_d, b_hh_d, b_out,
      pWihD, pWhhD, pEmbD, pWout,
      ctx_ws, h0_ws, outRecon);
}
